// AttributeContrastiveLoss_72541997629821
// MI455X (gfx1250) — matrix operand order repaired and checked by reading, not compiled, not run on hardware
//
#include <hip/hip_runtime.h>
#include <math.h>

// Problem constants (match reference)
#define BATCH 8192
#define DIM   512
#define NPROTO 16
#define NATTR  4
#define INV_TEMP (1.0f / 0.07f)

typedef __attribute__((ext_vector_type(16))) _Float16 v16h;
typedef __attribute__((ext_vector_type(8)))  _Float16 v8h;
typedef __attribute__((ext_vector_type(8)))  float    v8f;

// ---------------------------------------------------------------------------
// Kernel 1: row L2-normalize, emit f16 copy for WMMA, attribute code + pos flag
// ---------------------------------------------------------------------------
__global__ void k_normalize(const float* __restrict__ z,
                            const float* __restrict__ attrs,
                            _Float16* __restrict__ zh,
                            float* __restrict__ inv_norm,
                            int*   __restrict__ codes,
                            float* __restrict__ posf) {
  __shared__ float sm[256];
  const int row = blockIdx.x;
  const int t   = threadIdx.x;
  const float* zr = z + (size_t)row * DIM;
  float a = zr[t], b = zr[t + 256];
  sm[t] = a * a + b * b;
  __syncthreads();
  for (int s = 128; s > 0; s >>= 1) {
    if (t < s) sm[t] += sm[t + s];
    __syncthreads();
  }
  const float inv = 1.0f / fmaxf(sqrtf(sm[0]), 1e-12f);
  zh[(size_t)row * DIM + t]       = (_Float16)(a * inv);
  zh[(size_t)row * DIM + t + 256] = (_Float16)(b * inv);
  if (t == 0) {
    inv_norm[row] = inv;
    int c = 0;
#pragma unroll
    for (int k = 0; k < NATTR; ++k)
      c |= (attrs[row * NATTR + k] > 0.5f) ? (1 << (NATTR - 1 - k)) : 0;
    codes[row] = c;
    posf[row]  = (c == NPROTO - 1) ? 1.0f : 0.0f;
  }
}

// ---------------------------------------------------------------------------
// Kernel 2: prototypes = l2norm(segment-mean of z_norm); zero if empty segment
// ---------------------------------------------------------------------------
__global__ void k_proto(const float* __restrict__ z,
                        const float* __restrict__ inv_norm,
                        const int*   __restrict__ codes,
                        float* __restrict__ proto) {
  __shared__ float sm[256];
  const int p = blockIdx.x;
  const int t = threadIdx.x;
  float a0 = 0.f, a1 = 0.f;
  int cnt = 0;
  for (int r = 0; r < BATCH; ++r) {
    if (codes[r] == p) {
      const float inv = inv_norm[r];
      const float* zr = z + (size_t)r * DIM;
      a0 += zr[t] * inv;
      a1 += zr[t + 256] * inv;
      ++cnt;
    }
  }
  const float c = fmaxf((float)cnt, 1.0f);
  a0 /= c; a1 /= c;
  sm[t] = a0 * a0 + a1 * a1;
  __syncthreads();
  for (int s = 128; s > 0; s >>= 1) {
    if (t < s) sm[t] += sm[t + s];
    __syncthreads();
  }
  const float scale = (cnt > 0) ? (1.0f / fmaxf(sqrtf(sm[0]), 1e-12f)) : 0.0f;
  proto[(size_t)p * DIM + t]       = a0 * scale;
  proto[(size_t)p * DIM + t + 256] = a1 * scale;
}

// ---------------------------------------------------------------------------
// Kernel 3: prototype contrastive term (exact f32; tiny). One wave per row.
// ---------------------------------------------------------------------------
__global__ void k_proto_loss(const float* __restrict__ z,
                             const float* __restrict__ inv_norm,
                             const int*   __restrict__ codes,
                             const float* __restrict__ proto,
                             float* __restrict__ loss_acc) {
  const int gid  = blockIdx.x * blockDim.x + threadIdx.x;
  const int row  = gid >> 5;
  const int lane = threadIdx.x & 31;
  float acc[NPROTO];
#pragma unroll
  for (int p = 0; p < NPROTO; ++p) acc[p] = 0.f;
  const float inv = inv_norm[row];
  const float* zr = z + (size_t)row * DIM;
  for (int d = lane; d < DIM; d += 32) {
    const float zd = zr[d] * inv;
#pragma unroll
    for (int p = 0; p < NPROTO; ++p) acc[p] += zd * proto[p * DIM + d];
  }
#pragma unroll
  for (int p = 0; p < NPROTO; ++p) {
#pragma unroll
    for (int off = 16; off >= 1; off >>= 1)
      acc[p] += __shfl_xor(acc[p], off, 32);
  }
  if (lane == 0) {
    float e[NPROTO];
    float alls = 0.f;
#pragma unroll
    for (int p = 0; p < NPROTO; ++p) {
      e[p] = __expf(acc[p] * INV_TEMP);
      alls += e[p];
    }
    alls = fmaxf(alls, 1e-8f);
    const int c = codes[row];
    float pos = 0.f;
#pragma unroll
    for (int p = 0; p < NPROTO; ++p) pos = (p == c) ? e[p] : pos;
    atomicAdd(loss_acc, logf(pos / alls + 1e-8f));
  }
}

// ---------------------------------------------------------------------------
// A-fragment loader: row-major 16x32 f16 tile, contiguous K layout
//   elems 0..15 : K = kb + hi*16 + i
// p must already point at (row base + hi*16) within the row.
// ---------------------------------------------------------------------------
__device__ __forceinline__ v16h load_a_frag(const _Float16* __restrict__ p) {
  v8h lo = *(const v8h*)(p);
  v8h hi8 = *(const v8h*)(p + 8);
  return __builtin_shufflevector(lo, hi8, 0, 1, 2, 3, 4, 5, 6, 7,
                                 8, 9, 10, 11, 12, 13, 14, 15);
}

// ---------------------------------------------------------------------------
// Kernel 4: pairwise term via WMMA f16 GEMM (Z @ Z^T), fused exp + masked
// row-sum reduction. Each wave computes a 16-row x 64-col strip over K=512,
// software-pipelined: next K-step's A + 4 B fragments are prefetched before
// the current step's 4 WMMAs, so load latency overlaps matrix math.
// 8MB f16 operand set is fully L2-resident (192MB L2) -> no LDS staging.
// ---------------------------------------------------------------------------
__global__ void k_pairwise(const _Float16* __restrict__ zh,
                           const float* __restrict__ posf,
                           float* __restrict__ all_sims,
                           float* __restrict__ pos_sims) {
  const int lane = threadIdx.x & 31;
  const int wv   = threadIdx.x >> 5;   // 0..7
  const int l16  = lane & 15;
  const int hi   = lane >> 4;          // 0 or 1
  const int rowbase = blockIdx.x * 16;
  const int colbase = blockIdx.y * 512 + wv * 64;

  v8f acc[4] = {};

  // Fragment base pointers (lane-resolved)
  const _Float16* Ap  = zh + (size_t)(rowbase + l16) * DIM + hi * 16;
  const _Float16* Bp0 = zh + (size_t)(colbase +  0 + l16) * DIM + hi * 16;
  const _Float16* Bp1 = zh + (size_t)(colbase + 16 + l16) * DIM + hi * 16;
  const _Float16* Bp2 = zh + (size_t)(colbase + 32 + l16) * DIM + hi * 16;
  const _Float16* Bp3 = zh + (size_t)(colbase + 48 + l16) * DIM + hi * 16;

  // Prologue: fragments for kb = 0
  v16h a  = load_a_frag(Ap);
  v16h b0 = *(const v16h*)(Bp0);
  v16h b1 = *(const v16h*)(Bp1);
  v16h b2 = *(const v16h*)(Bp2);
  v16h b3 = *(const v16h*)(Bp3);

  for (int kb = 0; kb < DIM; kb += 32) {
    const int kn = (kb + 32) & (DIM - 1);  // wrap: last prefetch is harmless
    // Prefetch next K-step (issues before the waits on current fragments)
    v16h an  = load_a_frag(Ap + kn);
    v16h bn0 = *(const v16h*)(Bp0 + kn);
    v16h bn1 = *(const v16h*)(Bp1 + kn);
    v16h bn2 = *(const v16h*)(Bp2 + kn);
    v16h bn3 = *(const v16h*)(Bp3 + kn);

    acc[0] = __builtin_amdgcn_wmma_f32_16x16x32_f16(false, a, false, b0,
                                                    (short)0, acc[0], false, false);
    acc[1] = __builtin_amdgcn_wmma_f32_16x16x32_f16(false, a, false, b1,
                                                    (short)0, acc[1], false, false);
    acc[2] = __builtin_amdgcn_wmma_f32_16x16x32_f16(false, a, false, b2,
                                                    (short)0, acc[2], false, false);
    acc[3] = __builtin_amdgcn_wmma_f32_16x16x32_f16(false, a, false, b3,
                                                    (short)0, acc[3], false, false);

    a = an; b0 = bn0; b1 = bn1; b2 = bn2; b3 = bn3;
  }

  // D layout: elem r <-> row (rowbase + hi*8 + r), col (colbase + j*16 + l16)
  const int mbase = rowbase + hi * 8;
  float pm[8], rowsum[8], possum[8];
#pragma unroll
  for (int r = 0; r < 8; ++r) {
    pm[r] = posf[mbase + r];
    rowsum[r] = 0.f;
    possum[r] = 0.f;
  }
#pragma unroll
  for (int j = 0; j < 4; ++j) {
    const int n = colbase + j * 16 + l16;
    const float pc = posf[n];
#pragma unroll
    for (int r = 0; r < 8; ++r) {
      float e = __expf(acc[j][r] * INV_TEMP);
      e *= (mbase + r != n) ? 1.0f : 0.0f;  // exclude diagonal
      rowsum[r] += e;
      possum[r] += e * pc;
    }
  }
  // reduce across the 16 lanes holding the same output row
#pragma unroll
  for (int r = 0; r < 8; ++r) {
#pragma unroll
    for (int off = 8; off >= 1; off >>= 1) {
      rowsum[r] += __shfl_xor(rowsum[r], off, 32);
      possum[r] += __shfl_xor(possum[r], off, 32);
    }
  }
  if (l16 == 0) {
#pragma unroll
    for (int r = 0; r < 8; ++r) {
      atomicAdd(&all_sims[mbase + r], rowsum[r]);
      atomicAdd(&pos_sims[mbase + r], possum[r] * pm[r]);
    }
  }
}

// ---------------------------------------------------------------------------
// Kernel 5: finalize -> scalar loss
// ---------------------------------------------------------------------------
__global__ void k_finalize(const float* __restrict__ all_sims,
                           const float* __restrict__ pos_sims,
                           const float* __restrict__ posf,
                           const float* __restrict__ proto_loss,
                           float* __restrict__ out) {
  __shared__ float s_sum[256];
  __shared__ float s_cnt[256];
  const int t = threadIdx.x;
  float lsum = 0.f, lcnt = 0.f;
  for (int i = t; i < BATCH; i += 256) {
    if (posf[i] > 0.f && pos_sims[i] > 0.f) {
      lsum += -logf(pos_sims[i] / (all_sims[i] + 1e-8f) + 1e-8f);
      lcnt += 1.f;
    }
  }
  s_sum[t] = lsum; s_cnt[t] = lcnt;
  __syncthreads();
  for (int s = 128; s > 0; s >>= 1) {
    if (t < s) { s_sum[t] += s_sum[t + s]; s_cnt[t] += s_cnt[t + s]; }
    __syncthreads();
  }
  if (t == 0) {
    const float pl = -proto_loss[0] / (float)BATCH;
    const float nv = s_cnt[0];
    const float pw = s_sum[0] / fmaxf(nv, 1.0f);
    out[0] = (nv > 0.f) ? (0.7f * pl + 0.3f * pw) : pl;
  }
}

// ---------------------------------------------------------------------------
// Host launcher
// ---------------------------------------------------------------------------
extern "C" void kernel_launch(void* const* d_in, const int* in_sizes, int n_in,
                              void* d_out, int out_size, void* d_ws, size_t ws_size,
                              hipStream_t stream) {
  (void)in_sizes; (void)n_in; (void)out_size; (void)ws_size;
  const float* z     = (const float*)d_in[0];   // [8192, 512] f32
  const float* attrs = (const float*)d_in[1];   // [8192, 4]   f32
  float* out = (float*)d_out;                   // scalar f32

  // Workspace layout
  char* ws = (char*)d_ws;
  const size_t off_zh   = 0;                                    // 8192*512*2 = 8388608
  const size_t off_inv  = off_zh  + (size_t)BATCH * DIM * 2;    // 32768
  const size_t off_code = off_inv + (size_t)BATCH * 4;          // 32768
  const size_t off_posf = off_code + (size_t)BATCH * 4;         // 32768
  const size_t off_prot = off_posf + (size_t)BATCH * 4;         // 16*512*4 = 32768
  const size_t off_all  = off_prot + (size_t)NPROTO * DIM * 4;  // 32768
  const size_t off_pos  = off_all + (size_t)BATCH * 4;          // 32768
  const size_t off_pls  = off_pos + (size_t)BATCH * 4;          // 256

  _Float16* zh      = (_Float16*)(ws + off_zh);
  float*    invn    = (float*)(ws + off_inv);
  int*      codes   = (int*)(ws + off_code);
  float*    posf    = (float*)(ws + off_posf);
  float*    proto   = (float*)(ws + off_prot);
  float*    allsims = (float*)(ws + off_all);
  float*    possims = (float*)(ws + off_pos);
  float*    plsum   = (float*)(ws + off_pls);

  // Zero the accumulator region (all_sims | pos_sims | proto_loss_sum)
  hipMemsetAsync(ws + off_all, 0, (size_t)BATCH * 4 * 2 + 256, stream);

  k_normalize<<<BATCH, 256, 0, stream>>>(z, attrs, zh, invn, codes, posf);
  k_proto<<<NPROTO, 256, 0, stream>>>(z, invn, codes, proto);
  k_proto_loss<<<(BATCH * 32) / 256, 256, 0, stream>>>(z, invn, codes, proto, plsum);

  dim3 g4(BATCH / 16, BATCH / 512);  // (512, 16) blocks, 256 threads = 8 waves
  k_pairwise<<<g4, 256, 0, stream>>>(zh, posf, allsims, possims);

  k_finalize<<<1, 256, 0, stream>>>(allsims, possims, posf, plsum, out);
}
